// RNNModel_27118423507723
// MI455X (gfx1250) — compile-verified, matrix operand order checked by reading, not run on hardware
//
#include <hip/hip_runtime.h>
#include <hip/hip_bf16.h>

typedef unsigned short u16;
typedef __attribute__((ext_vector_type(16))) __bf16        v16bf;
typedef __attribute__((ext_vector_type(8)))  float         v8f;
typedef __attribute__((ext_vector_type(4)))  unsigned int  v4u;

__device__ __forceinline__ u16 f2bf(float f) {
    union { float f; unsigned u; } c; c.f = f;
    unsigned u = c.u;
    unsigned r = u + 0x7fffu + ((u >> 16) & 1u);   // round-to-nearest-even
    return (u16)(r >> 16);
}

// ---------------------------------------------------------------------------
// WMMA GEMM (wave per 16x16 tile): C[M,Np] = A(bf16 row-major) @ Bpacked(bf16)
// Bpacked (G=1): fragment-major [nt][kb][lane][16 halfwords] (512 u16/frag).
// Used for the small serial cell GEMMs (max wave parallelism, latency-bound).
// ---------------------------------------------------------------------------
__global__ void gemm_wmma_bf16(const u16* __restrict__ A, const u16* __restrict__ Bp,
                               float* __restrict__ C, const float* __restrict__ bias,
                               int Mt, int Nt, int KB, int lda, int ldc)
{
    int wave = blockIdx.x * (blockDim.x >> 5) + (threadIdx.x >> 5);
    int lane = threadIdx.x & 31;
    if (wave >= Mt * Nt) return;
    int mt = wave / Nt;
    int nt = wave % Nt;

    v8f acc = {};
    const u16* pa = A + (size_t)(mt * 16 + (lane & 15)) * lda + ((lane >> 4) << 3);
    const u16* pb = Bp + ((size_t)nt * KB) * 512 + lane * 16;

    for (int kb = 0; kb < KB; ++kb) {
        union { v4u u[2]; v16bf v; } a, b;
        a.u[0] = *(const v4u*)(pa);        // k = +0..+7 (of lane half)
        a.u[1] = *(const v4u*)(pa + 16);   // k = +16..+23
        b.u[0] = *(const v4u*)(pb);
        b.u[1] = *(const v4u*)(pb + 8);
        acc = __builtin_amdgcn_wmma_f32_16x16x32_bf16(false, a.v, false, b.v,
                                                      (short)0, acc, false, false);
        pa += 32;
        pb += 512;
    }

    int row0 = mt * 16 + ((lane >> 4) << 3);
    int col  = nt * 16 + (lane & 15);
    float bv = bias ? bias[col] : 0.0f;
#pragma unroll
    for (int v = 0; v < 8; ++v)
        C[(size_t)(row0 + v) * ldc + col] = acc[v] + bv;
}

// ---------------------------------------------------------------------------
// WMMA GEMM, 1x4 N-blocked (wave per 16x64 tile strip). A fragment reused by
// 4 independent WMMAs (no accumulator RAW chains). Bpacked with G=4 interleave:
// the 4 B fragments of one k-step are contiguous (2048 u16). Used for decoder.
// ---------------------------------------------------------------------------
__global__ void gemm_wmma_bf16_n4(const u16* __restrict__ A, const u16* __restrict__ Bp,
                                  float* __restrict__ C, const float* __restrict__ bias,
                                  int Mt, int Ng, int KB, int lda, int ldc, int Ncols)
{
    int wave = blockIdx.x * (blockDim.x >> 5) + (threadIdx.x >> 5);
    int lane = threadIdx.x & 31;
    if (wave >= Mt * Ng) return;
    int mt = wave / Ng;
    int ng = wave % Ng;

    v8f acc0 = {}, acc1 = {}, acc2 = {}, acc3 = {};
    const u16* pa = A + (size_t)(mt * 16 + (lane & 15)) * lda + ((lane >> 4) << 3);
    const u16* pb = Bp + ((size_t)ng * KB) * 2048 + lane * 16;

    for (int kb = 0; kb < KB; ++kb) {
        union { v4u u[2]; v16bf v; } a, b0, b1, b2, b3;
        a.u[0]  = *(const v4u*)(pa);
        a.u[1]  = *(const v4u*)(pa + 16);
        b0.u[0] = *(const v4u*)(pb);          b0.u[1] = *(const v4u*)(pb + 8);
        b1.u[0] = *(const v4u*)(pb + 512);    b1.u[1] = *(const v4u*)(pb + 520);
        b2.u[0] = *(const v4u*)(pb + 1024);   b2.u[1] = *(const v4u*)(pb + 1032);
        b3.u[0] = *(const v4u*)(pb + 1536);   b3.u[1] = *(const v4u*)(pb + 1544);
        acc0 = __builtin_amdgcn_wmma_f32_16x16x32_bf16(false, a.v, false, b0.v,
                                                       (short)0, acc0, false, false);
        acc1 = __builtin_amdgcn_wmma_f32_16x16x32_bf16(false, a.v, false, b1.v,
                                                       (short)0, acc1, false, false);
        acc2 = __builtin_amdgcn_wmma_f32_16x16x32_bf16(false, a.v, false, b2.v,
                                                       (short)0, acc2, false, false);
        acc3 = __builtin_amdgcn_wmma_f32_16x16x32_bf16(false, a.v, false, b3.v,
                                                       (short)0, acc3, false, false);
        pa += 32;
        pb += 2048;
    }

    int row0 = mt * 16 + ((lane >> 4) << 3);
    int colb = (ng * 4) * 16 + (lane & 15);
    v8f accs[4] = {acc0, acc1, acc2, acc3};
#pragma unroll
    for (int j = 0; j < 4; ++j) {
        int col = colb + j * 16;
        if (col >= Ncols) continue;
        float bv = bias ? bias[col] : 0.0f;
#pragma unroll
        for (int v = 0; v < 8; ++v)
            C[(size_t)(row0 + v) * ldc + col] = accs[j][v] + bv;
    }
}

// Pack f32 weight W(k,n) = src[k*sk + n*sn] into WMMA-B bf16 fragment layout,
// zero-padding K->KB*32, N->Nt*16. Fragment (nt,kb) stored at linear index
// ((nt/G)*KB + kb)*G + nt%G  (G-way n-tile interleave; G=1 => plain layout).
__global__ void pack_b_bf16(const float* __restrict__ src, u16* __restrict__ dst,
                            int K, int N, int KB, int Nt, int G, size_t sk, size_t sn)
{
    size_t id = (size_t)blockIdx.x * blockDim.x + threadIdx.x;
    size_t total = (size_t)Nt * KB * 512;
    if (id >= total) return;
    int r      = (int)(id & 511);
    size_t f   = id >> 9;
    int j      = (int)(f % G);
    size_t rest = f / G;
    int kbi    = (int)(rest % KB);
    int ntb    = (int)(rest / KB) * G + j;
    int lane = r >> 4, idx = r & 15;
    int n = ntb * 16 + (lane & 15);
    int k = kbi * 32 + ((idx >> 3) << 4) + ((lane >> 4) << 3) + (idx & 7);
    u16 v = 0;
    if (k < K && n < N) v = f2bf(src[(size_t)k * sk + (size_t)n * sn]);
    dst[id] = v;
}

__global__ void embed_bf16(const int* __restrict__ tok, const float* __restrict__ emb,
                           u16* __restrict__ x, int rows, int H)
{
    size_t id = (size_t)blockIdx.x * blockDim.x + threadIdx.x;
    if (id >= (size_t)rows * H) return;
    int r = (int)(id / H), j = (int)(id % H);
    x[id] = f2bf(emb[(size_t)tok[r] * H + j]);
}

__global__ void build_xh(const u16* __restrict__ x, const float* __restrict__ h,
                         u16* __restrict__ xh, int B, int H, int ldxh)
{
    int H2 = 2 * H;
    size_t id = (size_t)blockIdx.x * blockDim.x + threadIdx.x;
    if (id >= (size_t)B * H2) return;
    int b = (int)(id / H2), j = (int)(id % H2);
    u16 v = (j < H) ? x[(size_t)b * H + j] : f2bf(h[(size_t)b * H + (j - H)]);
    xh[(size_t)b * ldxh + j] = v;
}

__device__ __forceinline__ float actf(int a, float x) {
    if (a == 0) return tanhf(x);
    if (a == 1) return fmaxf(x, 0.0f);
    if (a == 2) return 1.0f / (1.0f + __expf(-x));
    return x;
}

// s = sp + sigmoid(c) * (act(h) - sp); write f32 + padded-bf16 state; accumulate mean sum
__global__ void cell_node(const float* __restrict__ pre, int ldp,
                          const float* __restrict__ sp,
                          float* __restrict__ sf, u16* __restrict__ sb, int ldsb,
                          float* __restrict__ sum, int act, int accmode, int B, int H)
{
    size_t id = (size_t)blockIdx.x * blockDim.x + threadIdx.x;
    if (id >= (size_t)B * H) return;
    int b = (int)(id / H), j = (int)(id % H);
    float c  = pre[(size_t)b * ldp + j];
    float hh = pre[(size_t)b * ldp + H + j];
    float g  = 1.0f / (1.0f + __expf(-c));
    float p  = sp[id];
    float s  = p + g * (actf(act, hh) - p);
    sf[id] = s;
    sb[(size_t)b * ldsb + j] = f2bf(s);
    if (accmode == 1)      sum[id] = s;
    else if (accmode == 2) sum[id] += s;
}

__global__ void cell_mean(const float* __restrict__ sum, float* __restrict__ h,
                          u16* __restrict__ outrow, int ldo, int B, int H)
{
    size_t id = (size_t)blockIdx.x * blockDim.x + threadIdx.x;
    if (id >= (size_t)B * H) return;
    int b = (int)(id / H), j = (int)(id % H);
    float m = sum[id] * 0.125f;
    h[id] = m;
    outrow[(size_t)b * ldo + j] = f2bf(m);
}

__global__ void fill_zero_u16(u16* __restrict__ p, size_t n)
{
    size_t id = (size_t)blockIdx.x * blockDim.x + threadIdx.x;
    if (id < n) p[id] = 0;
}

__global__ void copy_f32(const float* __restrict__ s, float* __restrict__ d, size_t n)
{
    size_t id = (size_t)blockIdx.x * blockDim.x + threadIdx.x;
    if (id < n) d[id] = s[id];
}

__global__ void log_softmax_rows(float* __restrict__ out, int V)
{
    __shared__ float red[256];
    float* p = out + (size_t)blockIdx.x * V;
    int tid = threadIdx.x;
    float mx = -3.4e38f;
    for (int j = tid; j < V; j += 256) mx = fmaxf(mx, p[j]);
    red[tid] = mx; __syncthreads();
    for (int s = 128; s > 0; s >>= 1) {
        if (tid < s) red[tid] = fmaxf(red[tid], red[tid + s]);
        __syncthreads();
    }
    mx = red[0]; __syncthreads();
    float sm = 0.0f;
    for (int j = tid; j < V; j += 256) sm += __expf(p[j] - mx);
    red[tid] = sm; __syncthreads();
    for (int s = 128; s > 0; s >>= 1) {
        if (tid < s) red[tid] += red[tid + s];
        __syncthreads();
    }
    float lse = mx + __logf(red[0]);
    __syncthreads();
    for (int j = tid; j < V; j += 256) p[j] -= lse;
}

// ---------------------------------------------------------------------------
static inline int gblk(size_t n, int t) { return (int)((n + (size_t)t - 1) / t); }

extern "C" void kernel_launch(void* const* d_in, const int* in_sizes, int n_in,
                              void* d_out, int out_size, void* d_ws, size_t ws_size,
                              hipStream_t stream)
{
    (void)in_sizes; (void)n_in; (void)out_size; (void)ws_size;

    const int T = 70, B = 64, H = 850, H2 = 1700, V = 10000;
    const int Hp = 864, H2p = 1728;          // K padded to multiples of 32
    const int KB0 = H2p / 32;                // 54 (W0 GEMM)
    const int KBn = Hp / 32;                 // 27 (node / decoder GEMMs)
    const int NtC = H2p / 16;                // 108 (cell GEMM N tiles)
    const int NtV = 628;                     // decoder N tiles padded to mult of 4 (625 -> 628)
    const int NgV = NtV / 4;                 // 157 groups of 4

    const int*   tokens = (const int*)d_in[0];
    const float* h0     = (const float*)d_in[1];
    const float* emb_w  = (const float*)d_in[2];
    const float* dec_b  = (const float*)d_in[3];
    const float* W0_1   = (const float*)d_in[4];
    const float* Ws_1   = (const float*)d_in[5];
    const float* W0_2   = (const float*)d_in[6];
    const float* Ws_2   = (const float*)d_in[7];
    float* out = (float*)d_out;

    // workspace bump allocator (256B aligned)
    char*  base = (char*)d_ws;
    size_t off  = 0;
    auto alloc = [&](size_t bytes) -> char* {
        char* p = base + off;
        off += (bytes + 255) & ~(size_t)255;
        return p;
    };
    u16*   w0p1 = (u16*)  alloc((size_t)H2p * H2p * 2);
    u16*   w0p2 = (u16*)  alloc((size_t)H2p * H2p * 2);
    u16*   wsp1 = (u16*)  alloc((size_t)8 * Hp * H2p * 2);
    u16*   wsp2 = (u16*)  alloc((size_t)8 * Hp * H2p * 2);
    u16*   embp = (u16*)  alloc((size_t)Hp * NtV * 16 * 2);
    u16*   x1   = (u16*)  alloc((size_t)T * B * H * 2);
    u16*   out1 = (u16*)  alloc((size_t)T * B * H * 2);
    u16*   out2 = (u16*)  alloc((size_t)T * B * Hp * 2);
    u16*   xh   = (u16*)  alloc((size_t)B * H2p * 2);
    float* stf  = (float*)alloc((size_t)9 * B * H * 4);
    u16*   stb  = (u16*)  alloc((size_t)9 * B * Hp * 2);
    float* pre  = (float*)alloc((size_t)B * H2p * 4);
    float* hbuf = (float*)alloc((size_t)B * H * 4);
    float* sum  = (float*)alloc((size_t)B * H * 4);

    // ---- pack weights to bf16 WMMA-B fragments ----
    {
        size_t n = (size_t)NtC * KB0 * 512;
        pack_b_bf16<<<gblk(n, 256), 256, 0, stream>>>(W0_1, w0p1, H2, H2, KB0, NtC, 1, H2, 1);
        pack_b_bf16<<<gblk(n, 256), 256, 0, stream>>>(W0_2, w0p2, H2, H2, KB0, NtC, 1, H2, 1);
    }
    {
        size_t n = (size_t)NtC * KBn * 512;
        for (int i = 0; i < 8; ++i) {
            pack_b_bf16<<<gblk(n, 256), 256, 0, stream>>>(
                Ws_1 + (size_t)i * H * H2, wsp1 + (size_t)i * Hp * H2p, H, H2, KBn, NtC, 1, H2, 1);
            pack_b_bf16<<<gblk(n, 256), 256, 0, stream>>>(
                Ws_2 + (size_t)i * H * H2, wsp2 + (size_t)i * Hp * H2p, H, H2, KBn, NtC, 1, H2, 1);
        }
    }
    {   // decoder B = emb_w^T : W(k,n) = emb_w[n*H + k]; G=4 interleave for n4 kernel
        size_t n = (size_t)NtV * KBn * 512;
        pack_b_bf16<<<gblk(n, 256), 256, 0, stream>>>(emb_w, embp, H, V, KBn, NtV, 4, 1, H);
    }

    // ---- embedding lookup + state init ----
    embed_bf16<<<gblk((size_t)T * B * H, 256), 256, 0, stream>>>(tokens, emb_w, x1, T * B, H);
    fill_zero_u16<<<gblk((size_t)9 * B * Hp, 256), 256, 0, stream>>>(stb, (size_t)9 * B * Hp);
    fill_zero_u16<<<gblk((size_t)B * H2p, 256), 256, 0, stream>>>(xh, (size_t)B * H2p);
    fill_zero_u16<<<gblk((size_t)T * B * Hp, 256), 256, 0, stream>>>(out2, (size_t)T * B * Hp);
    copy_f32<<<gblk((size_t)B * H, 256), 256, 0, stream>>>(h0, hbuf, (size_t)B * H);

    // DARTS_V1 genotype: activation enum {0=tanh,1=relu,2=sigmoid,3=identity}
    const int pred[8] = {0, 1, 1, 1, 2, 5, 3, 5};
    const int acts[8] = {2, 1, 1, 3, 0, 2, 0, 1};

    const int cellBlocks = (4 * NtC + 7) / 8;          // 432 tiles, 8 waves/block
    const size_t nBH  = (size_t)B * H;
    const size_t nBH2 = (size_t)B * H2;

    for (int layer = 0; layer < 2; ++layer) {
        const u16* xin  = layer ? out1 : x1;
        const u16* w0p  = layer ? w0p2 : w0p1;
        const u16* wsp  = layer ? wsp2 : wsp1;
        u16*       outb = layer ? out2 : out1;
        const int  ldo  = layer ? Hp : H;

        for (int t = 0; t < T; ++t) {
            build_xh<<<gblk(nBH2, 256), 256, 0, stream>>>(
                xin + (size_t)t * B * H, hbuf, xh, B, H, H2p);
            gemm_wmma_bf16<<<cellBlocks, 256, 0, stream>>>(
                xh, w0p, pre, nullptr, 4, NtC, KB0, H2p, H2p);
            cell_node<<<gblk(nBH, 256), 256, 0, stream>>>(
                pre, H2p, hbuf, stf, stb, Hp, sum, /*tanh*/0, /*no acc*/0, B, H);
            for (int i = 0; i < 8; ++i) {
                gemm_wmma_bf16<<<cellBlocks, 256, 0, stream>>>(
                    stb + (size_t)pred[i] * B * Hp, wsp + (size_t)i * Hp * H2p,
                    pre, nullptr, 4, NtC, KBn, Hp, H2p);
                cell_node<<<gblk(nBH, 256), 256, 0, stream>>>(
                    pre, H2p, stf + (size_t)pred[i] * B * H,
                    stf + (size_t)(i + 1) * B * H, stb + (size_t)(i + 1) * B * Hp, Hp,
                    sum, acts[i], (i == 0) ? 1 : 2, B, H);
            }
            cell_mean<<<gblk(nBH, 256), 256, 0, stream>>>(
                sum, hbuf, outb + (size_t)t * B * ldo, ldo, B, H);
        }
    }

    // ---- tied decoder GEMM (1x4 N-blocked) into d_out, then in-place log-softmax ----
    {
        int Mt = (T * B) / 16;                           // 280
        int waves = Mt * NgV;                            // 43,960
        int blocks = (waves + 7) / 8;
        gemm_wmma_bf16_n4<<<blocks, 256, 0, stream>>>(
            out2, embp, out, dec_b, Mt, NgV, KBn, Hp, V, V);
    }
    log_softmax_rows<<<T * B, 256, 0, stream>>>(out, V);
}
